// MultiHeadLegendreGraphConvLayer_84224308675155
// MI455X (gfx1250) — matrix operand order repaired and checked by reading, not compiled, not run on hardware
//
#include <hip/hip_runtime.h>

typedef __attribute__((ext_vector_type(16))) __bf16 v16bf;
typedef __attribute__((ext_vector_type(8)))  float  v8f;

#define NB 8
#define NN 2048
#define FF 128
#define KP1 5
#define PROPF 640   // (K+1)*FF
#define OUTF 256

union V16U { uint4 q[2]; v16bf v; __bf16 h[16]; };

__device__ __forceinline__ unsigned pkbf(float a, float b) {
    union { __bf16 h[2]; unsigned u; } u;
    u.h[0] = (__bf16)a; u.h[1] = (__bf16)b;
    return u.u;
}

// ---------------------------------------------------------------- d[b,n] = rsqrt(1 + rowsum(adj))
__global__ __launch_bounds__(256)
void rownorm_kernel(const float* __restrict__ adj, float* __restrict__ d) {
    int row = blockIdx.x;                    // b*NN + n
    const float* p = adj + (size_t)row * NN;
    float s = 0.f;
    for (int i = threadIdx.x; i < NN; i += 256) s += p[i];
    __shared__ float red[256];
    red[threadIdx.x] = s; __syncthreads();
    for (int off = 128; off > 0; off >>= 1) {
        if (threadIdx.x < off) red[threadIdx.x] += red[threadIdx.x + off];
        __syncthreads();
    }
    if (threadIdx.x == 0) d[row] = rsqrtf(red[0] + 1.0f);
}

// ------------------------------------------- xT[b][f][m] = bf16(x[b][m][f]); colsum[b][f] += sum_m x
__global__ __launch_bounds__(256)
void xprep_kernel(const float* __restrict__ x, __bf16* __restrict__ xT,
                  float* __restrict__ colsum) {
    __shared__ float t[32 * 133];
    int b = blockIdx.y;
    int m0 = blockIdx.x * 32;
    {   // coalesced read of 32x128 tile
        int r = threadIdx.x >> 3;
        int fs = (threadIdx.x & 7) * 16;
        const float* src = x + (((size_t)b * NN) + m0 + r) * FF + fs;
        float* dst = t + r * 133 + fs;
        #pragma unroll
        for (int i = 0; i < 16; ++i) dst[i] = src[i];
    }
    __syncthreads();
    {   // transposed bf16 write, 32B per thread
        int f = threadIdx.x >> 1, half = threadIdx.x & 1;
        V16U up;
        #pragma unroll
        for (int i = 0; i < 16; ++i) up.h[i] = (__bf16)t[(half * 16 + i) * 133 + f];
        uint4* dq = (uint4*)(xT + (((size_t)b * FF) + f) * NN + m0 + half * 16);
        dq[0] = up.q[0]; dq[1] = up.q[1];
    }
    if (threadIdx.x < FF) {
        float s = 0.f;
        #pragma unroll
        for (int m = 0; m < 32; ++m) s += t[m * 133 + threadIdx.x];
        atomicAdd(&colsum[b * FF + threadIdx.x], s);
    }
}

// -------------------------------- Wcomb = W_out @ W_heads_flat (bf16), bcomb = W_out @ bh + b_out
__global__ __launch_bounds__(256)
void wprep_kernel(const float* __restrict__ w_heads, const float* __restrict__ b_heads,
                  const float* __restrict__ w_out, const float* __restrict__ b_out,
                  __bf16* __restrict__ wcomb, float* __restrict__ bcomb) {
    int o = blockIdx.x;                       // 0..255
    const float* wo = w_out + (size_t)o * OUTF;
    for (int c = threadIdx.x; c < PROPF; c += 256) {
        float s = 0.f;
        for (int p = 0; p < OUTF; ++p) s += wo[p] * w_heads[(size_t)p * PROPF + c];
        wcomb[(size_t)o * PROPF + c] = (__bf16)s;
    }
    if (threadIdx.x == 0) {
        float s = b_out[o];
        for (int p = 0; p < OUTF; ++p) s += wo[p] * b_heads[p];
        bcomb[o] = s;
    }
}

// ---------------------------------------------------------------- fused powers + WMMA + Legendre
// block: 256 thr = 8 waves; covers 32 rows (2 n-tiles) x 128 f x j=1..4
// wave (wid): tIdx = wid>>2 selects n-tile, fQuad = wid&3 selects 32-col f slab
// Double-buffered LDS: stage chunk i+1 while WMMAs consume chunk i; ONE barrier per chunk.
__global__ __launch_bounds__(256)
void legendre_spmm_kernel(const float* __restrict__ adj, const float* __restrict__ x,
                          const float* __restrict__ d, const float* __restrict__ colsum,
                          const __bf16* __restrict__ xT, __bf16* __restrict__ propbf) {
    __shared__ __align__(32) __bf16 ldsX[2][FF * 40];          // x tile [128 f][32 m] pad->40
    __shared__ __align__(32) __bf16 ldsA[2][4 * 2 * 32 * 16];  // [j][tile][lane][16] frag layout

    const int tid  = threadIdx.x;
    const int lane = tid & 31;
    const int wid  = tid >> 5;
    const int tIdx = wid >> 2;       // 0..1
    const int fQuad = wid & 3;       // 0..3
    const int b  = blockIdx.y;
    const int n0 = blockIdx.x * 32;

    // staging-role constants
    const int sf   = tid >> 1;       // x-stage: f row
    const int shalf = tid & 1;       // x-stage: which 16-m half
    const int sc   = (tid & 15) * 2; // A-stage: column pair within chunk
    const int sr   = tid >> 4;       // A-stage: row 0..15
    const int laneT = ((sc >> 4) << 4) + sr;
    const int slotW = (sc & 15) >> 1;

    v8f acc[4][2];
    #pragma unroll
    for (int j = 0; j < 4; ++j)
        #pragma unroll
        for (int ft = 0; ft < 2; ++ft) acc[j][ft] = (v8f)0.f;

    // ---- stage helper (inlined twice) ----
    auto stage = [&](int buf, int mc) {
        // x tile -> LDS (B-fragment friendly rows of 32 bf16)
        {
            const uint4* src = (const uint4*)(xT + (((size_t)b * FF) + sf) * NN + mc + shalf * 16);
            uint4* dst = (uint4*)(&ldsX[buf][sf * 40 + shalf * 16]);
            dst[0] = src[0]; dst[1] = src[1];
        }
        // g = a*dm powers -> bf16 A fragments in LDS
        {
            float2 dm = *(const float2*)(d + b * NN + mc + sc);
            unsigned* abase = (unsigned*)&ldsA[buf][0];
            #pragma unroll
            for (int t = 0; t < 2; ++t) {
                int n = n0 + t * 16 + sr;
                const float* arow = adj + (((size_t)b * NN) + n) * NN + mc + sc;
                float2 a = *(const float2*)arow;
                // prefetch adj two chunks (64 cols) ahead (streaming HBM input)
                __builtin_prefetch(arow + 64, 0, 1);
                float g0 = a.x * dm.x, g1 = a.y * dm.y;
                float g20 = g0 * g0,  g21 = g1 * g1;
                float g30 = g20 * g0, g31 = g21 * g1;
                float g40 = g20 * g20, g41 = g21 * g21;
                unsigned* q = abase + (t * 32 + laneT) * 8 + slotW;
                q[0]          = pkbf(g0,  g1);    // j=1
                q[2 * 32 * 8] = pkbf(g20, g21);   // j=2
                q[4 * 32 * 8] = pkbf(g30, g31);   // j=3
                q[6 * 32 * 8] = pkbf(g40, g41);   // j=4
            }
        }
    };

    stage(0, 0);
    __syncthreads();

    int p = 0;
    for (int mc = 0; mc < NN; mc += 32, p ^= 1) {
        if (mc + 32 < NN) stage(p ^ 1, mc + 32);   // overlap staging with WMMA
        // ---- fragments + WMMA from buffer p ----
        {
            v16bf aF[4];
            #pragma unroll
            for (int j = 0; j < 4; ++j)
                aF[j] = *(const v16bf*)(&ldsA[p][((j * 2 + tIdx) * 32 + lane) * 16]);
            #pragma unroll
            for (int ft = 0; ft < 2; ++ft) {
                int f  = fQuad * 32 + ft * 16 + (lane & 15);
                int mo = (lane >> 4) * 16;
                V16U bu;
                const uint4* bp = (const uint4*)(&ldsX[p][f * 40 + mo]);
                bu.q[0] = bp[0]; bu.q[1] = bp[1];
                #pragma unroll
                for (int j = 0; j < 4; ++j)
                    acc[j][ft] = __builtin_amdgcn_wmma_f32_16x16x32_bf16(
                        false, aF[j], false, bu.v, (short)0, acc[j][ft], false, false);
            }
        }
        __syncthreads();
    }

    // ---- epilogue: d_n^j scaling, +I diagonal correction, Legendre coeffs, bf16 prop store ----
    const int colLane = lane & 15;
    const int hhi = lane >> 4;
    #pragma unroll
    for (int ft = 0; ft < 2; ++ft) {
        int fg = fQuad * 32 + ft * 16 + colLane;
        float cs = colsum[b * FF + fg];
        #pragma unroll
        for (int e = 0; e < 8; ++e) {
            int n = n0 + tIdx * 16 + e + 8 * hhi;
            float dn  = d[b * NN + n];
            float ann = adj[(((size_t)b * NN) + n) * NN + n];
            float xv  = x[(((size_t)b * NN) + n) * FF + fg];
            float gd = ann * dn, gp = (ann + 1.f) * dn;
            float S[5]; S[0] = cs;
            float dnp = dn, gdj = gd, gpj = gp;
            #pragma unroll
            for (int j = 1; j <= 4; ++j) {
                float M = acc[j - 1][ft][e];
                S[j] = dnp * (M + (gpj - gdj) * xv);
                dnp *= dn; gdj *= gd; gpj *= gp;
            }
            float p0 = S[0];
            float p1 = S[1];
            float p2 = 1.5f * S[2] - 0.5f * S[0];
            float p3 = 2.5f * S[3] - 1.5f * S[1];
            float p4 = 4.375f * S[4] - 3.75f * S[2] + 0.375f * S[0];
            __bf16* po = propbf + (((size_t)b * NN) + n) * PROPF + fg;
            po[0 * FF] = (__bf16)p0;
            po[1 * FF] = (__bf16)p1;
            po[2 * FF] = (__bf16)p2;
            po[3 * FF] = (__bf16)p3;
            po[4 * FF] = (__bf16)p4;
        }
    }
}

// ---------------------------------------------------------------- out = prop @ Wcomb^T + bcomb
__global__ __launch_bounds__(256)
void out_gemm_kernel(const __bf16* __restrict__ propbf, const __bf16* __restrict__ wcomb,
                     const float* __restrict__ bcomb, float* __restrict__ out) {
    const int tid  = threadIdx.x;
    const int lane = tid & 31;
    const int wid  = tid >> 5;
    const int tIdx = wid >> 2;      // row tile 0..1
    const int cQuad = wid & 3;      // 64-col slab
    const int rows0 = blockIdx.x * 32;

    v8f acc[4];
    #pragma unroll
    for (int nt = 0; nt < 4; ++nt) acc[nt] = (v8f)0.f;

    const int r = lane & 15;
    const int h = lane >> 4;
    const int row = rows0 + tIdx * 16 + r;

    for (int kc = 0; kc < PROPF; kc += 32) {
        V16U au;
        const uint4* ap = (const uint4*)(propbf + (size_t)row * PROPF + kc + 16 * h);
        au.q[0] = ap[0]; au.q[1] = ap[1];   // one 16-elem run: k0+16h
        #pragma unroll
        for (int nt = 0; nt < 4; ++nt) {
            int ncol = cQuad * 64 + nt * 16 + (lane & 15);
            int ks = kc + 16 * (lane >> 4);
            const v16bf bv = *(const v16bf*)(wcomb + (size_t)ncol * PROPF + ks);
            acc[nt] = __builtin_amdgcn_wmma_f32_16x16x32_bf16(
                false, au.v, false, bv, (short)0, acc[nt], false, false);
        }
    }
    #pragma unroll
    for (int nt = 0; nt < 4; ++nt) {
        int col = cQuad * 64 + nt * 16 + (lane & 15);
        float bb = bcomb[col];
        #pragma unroll
        for (int e = 0; e < 8; ++e) {
            int rr = rows0 + tIdx * 16 + e + 8 * (lane >> 4);
            out[(size_t)rr * OUTF + col] = acc[nt][e] + bb;
        }
    }
}

extern "C" void kernel_launch(void* const* d_in, const int* in_sizes, int n_in,
                              void* d_out, int out_size, void* d_ws, size_t ws_size,
                              hipStream_t stream) {
    const float* x       = (const float*)d_in[0];
    const float* adj     = (const float*)d_in[1];
    const float* w_heads = (const float*)d_in[2];
    const float* b_heads = (const float*)d_in[3];
    const float* w_out   = (const float*)d_in[4];
    const float* b_out   = (const float*)d_in[5];
    float* out = (float*)d_out;

    char* ws = (char*)d_ws;
    size_t off = 0;
    float*  d_deg  = (float*)(ws + off); off += (size_t)NB * NN * 4;          // 64 KB
    float*  colsum = (float*)(ws + off); off += (size_t)NB * FF * 4;          // 4 KB
    float*  bcomb  = (float*)(ws + off); off += (size_t)OUTF * 4;             // 1 KB
    off = (off + 255) & ~(size_t)255;
    __bf16* xT     = (__bf16*)(ws + off); off += (size_t)NB * FF * NN * 2;    // 4 MB
    __bf16* wcomb  = (__bf16*)(ws + off); off += (size_t)OUTF * PROPF * 2;    // 320 KB
    off = (off + 255) & ~(size_t)255;
    __bf16* propbf = (__bf16*)(ws + off); off += (size_t)NB * NN * PROPF * 2; // 20 MB

    hipMemsetAsync(colsum, 0, (size_t)NB * FF * 4, stream);
    rownorm_kernel<<<NB * NN, 256, 0, stream>>>(adj, d_deg);
    xprep_kernel<<<dim3(NN / 32, NB), 256, 0, stream>>>(x, xT, colsum);
    wprep_kernel<<<OUTF, 256, 0, stream>>>(w_heads, b_heads, w_out, b_out, wcomb, bcomb);
    legendre_spmm_kernel<<<dim3(NN / 32, NB), 256, 0, stream>>>(adj, x, d_deg, colsum, xT, propbf);
    out_gemm_kernel<<<(NB * NN) / 32, 256, 0, stream>>>(propbf, wcomb, bcomb, out);
}
